// PointNet_72945724555783
// MI455X (gfx1250) — compile-verified, matrix operand order checked by reading, not run on hardware
//
#include <hip/hip_runtime.h>
#include <hip/hip_bf16.h>

// ---------------- problem constants (from reference) ----------------
#define BB   8
#define NN   4096
#define MM1  820
#define MM2  205
#define KNBR 64
#define RR1  0.2f
#define RR2  0.4f
#define EPSBN 1e-5f

#define ROWS1 (BB*MM1*KNBR)   // 419840
#define ROWS2 (BB*MM2*KNBR)   // 104960

typedef __attribute__((ext_vector_type(16))) _Float16 v16h;
typedef __attribute__((ext_vector_type(8)))  _Float16 v8h;
typedef __attribute__((ext_vector_type(8)))  float    v8f;

// ======================================================================
// Farthest point sampling: one workgroup per cloud, dist array in LDS.
// ======================================================================
__global__ void fps_kernel(const float* __restrict__ pos, int n, int M,
                           int* __restrict__ outIdx) {
  __shared__ float dmin[4096];
  __shared__ float red[256];
  __shared__ int   redi[256];
  __shared__ float cp[3];
  const int b = blockIdx.x;
  const int t = threadIdx.x;
  const float* P = pos + (size_t)b * n * 3;
  const float x0 = P[0], y0 = P[1], z0 = P[2];
  for (int i = t; i < n; i += blockDim.x) {
    float dx = P[i*3]-x0, dy = P[i*3+1]-y0, dz = P[i*3+2]-z0;
    dmin[i] = dx*dx + dy*dy + dz*dz;
  }
  if (t == 0) outIdx[(size_t)b*M] = 0;
  __syncthreads();
  for (int m = 1; m < M; ++m) {
    float best = -1.0f; int bi = n;
    for (int i = t; i < n; i += blockDim.x) {
      float d = dmin[i];
      if (d > best) { best = d; bi = i; }      // strict > keeps lowest index
    }
    red[t] = best; redi[t] = bi;
    __syncthreads();
    for (int s = blockDim.x >> 1; s > 0; s >>= 1) {
      if (t < s) {
        if (red[t+s] > red[t] || (red[t+s] == red[t] && redi[t+s] < redi[t])) {
          red[t] = red[t+s]; redi[t] = redi[t+s];
        }
      }
      __syncthreads();
    }
    if (t == 0) {
      int sel = redi[0];
      outIdx[(size_t)b*M + m] = sel;
      cp[0] = P[sel*3]; cp[1] = P[sel*3+1]; cp[2] = P[sel*3+2];
    }
    __syncthreads();
    float cx = cp[0], cy = cp[1], cz = cp[2];
    for (int i = t; i < n; i += blockDim.x) {
      float dx = P[i*3]-cx, dy = P[i*3+1]-cy, dz = P[i*3+2]-cz;
      float d = dx*dx + dy*dy + dz*dz;
      if (d < dmin[i]) dmin[i] = d;
    }
    __syncthreads();
  }
}

// ======================================================================
// gather center positions
// ======================================================================
__global__ void gather_pos(const float* __restrict__ pos, const int* __restrict__ idx,
                           int n, int M, int total, float* __restrict__ out) {
  int g = blockIdx.x*blockDim.x + threadIdx.x;
  if (g >= total) return;
  int b = g / M;
  int j = idx[g];
  const float* p = pos + ((size_t)b*n + j)*3;
  out[(size_t)g*3+0] = p[0];
  out[(size_t)g*3+1] = p[1];
  out[(size_t)g*3+2] = p[2];
}

// ======================================================================
// radius-ball query: per-center thread, sorted top-KNBR insertion
// ======================================================================
__global__ void radius_kernel(const float* __restrict__ pos_all, const float* __restrict__ pos_c,
                              int n, int M, float r2, int total,
                              int* __restrict__ nidx, int* __restrict__ valid) {
  int g = blockIdx.x*blockDim.x + threadIdx.x;
  if (g >= total) return;
  int b = g / M;
  const float* P = pos_all + (size_t)b*n*3;
  float cx = pos_c[(size_t)g*3], cy = pos_c[(size_t)g*3+1], cz = pos_c[(size_t)g*3+2];
  float bd[KNBR]; int bi[KNBR]; int cnt = 0;
  for (int i = 0; i < n; ++i) {
    float dx = P[i*3]-cx, dy = P[i*3+1]-cy, dz = P[i*3+2]-cz;
    float d = dx*dx + dy*dy + dz*dz;
    if (d > r2) continue;
    if (cnt == KNBR && d >= bd[KNBR-1]) continue;
    int j = (cnt < KNBR) ? cnt : KNBR-1;
    while (j > 0 && bd[j-1] > d) { bd[j] = bd[j-1]; bi[j] = bi[j-1]; --j; }
    bd[j] = d; bi[j] = i;
    if (cnt < KNBR) ++cnt;
  }
  for (int k = 0; k < KNBR; ++k) {
    nidx [(size_t)g*KNBR + k] = (k < cnt) ? bi[k] : 0;
    valid[(size_t)g*KNBR + k] = (k < cnt) ? 1 : 0;
  }
}

// ======================================================================
// build msg = cat([x_j, p_j - p_center]) rows, zero-padded to Cpad stride
// ======================================================================
__global__ void build_msg(const float* __restrict__ x, const float* __restrict__ pos,
                          const float* __restrict__ pos_c, const int* __restrict__ nidx,
                          int F, int n, int M, long rows, int Cpad,
                          float* __restrict__ out) {
  long row = (long)blockIdx.x*blockDim.x + threadIdx.x;
  if (row >= rows) return;
  long cen = row >> 6;              // / KNBR
  int  b   = (int)(cen / M);
  int  j   = nidx[row];
  const float* xp = x + ((size_t)b*n + j)*F;
  const float* pj = pos + ((size_t)b*n + j)*3;
  const float* pc = pos_c + (size_t)cen*3;
  float* o = out + row*(size_t)Cpad;
  for (int c = 0; c < F; ++c) o[c] = xp[c];
  o[F+0] = pj[0]-pc[0]; o[F+1] = pj[1]-pc[1]; o[F+2] = pj[2]-pc[2];
  for (int c = F+3; c < Cpad; ++c) o[c] = 0.f;
}

// ======================================================================
// pad W[Cin,Cout] -> Wp[Kpad,Cout] with zero rows beyond Cin
// ======================================================================
__global__ void pad_w(const float* __restrict__ W, float* __restrict__ Wp,
                      int Cin, int Cout, long total) {
  long g = (long)blockIdx.x*blockDim.x + threadIdx.x;
  if (g >= total) return;
  long k = g / Cout; int c = (int)(g % Cout);
  Wp[g] = (k < Cin) ? W[k*(size_t)Cout + c] : 0.f;
}

// ======================================================================
// WMMA GEMM: Y[R,Cout] = relu(X[R,Kpad] @ W[Kpad,Cout] + bias)
// Kpad is a multiple of 32 (producers zero-pad), so there are NO K
// guards: every K step is the clean path. 256 thr = 8 waves (4x2);
// block tile 128x64; each wave owns 2x2 16x16 tiles -> 4 v_wmma/K-step.
// A row-major in LDS, B transposed in LDS: fragment = 2x ds_load_b128.
// ======================================================================
__global__ void __launch_bounds__(256)
gemm_bias_relu_wmma(const float* __restrict__ X, const float* __restrict__ W,
                    const float* __restrict__ bias, float* __restrict__ Y,
                    int R, int Kpad, int Cout) {
  __shared__ _Float16 As [128][40];   // [row][k]   stride 80B (conflict-free 16B reads)
  __shared__ _Float16 Bst[ 64][40];   // [col][k]   transposed W tile
  const int tid  = threadIdx.x;
  const int lane = tid & 31;
  const int wave = tid >> 5;
  const int wm = wave & 3;            // 0..3 -> 32-row slice
  const int wn = wave >> 2;           // 0..1 -> 32-col slice
  const int rowBase = blockIdx.x * 128;
  const int colBase = blockIdx.y * 64;
  const bool fullRows = (rowBase + 128 <= R);

  const v8f vzero = {0.f,0.f,0.f,0.f,0.f,0.f,0.f,0.f};
  v8f acc[2][2];
  acc[0][0] = vzero; acc[0][1] = vzero; acc[1][0] = vzero; acc[1][1] = vzero;

  const int Ksteps = Kpad >> 5;
  for (int ks = 0; ks < Ksteps; ++ks) {
    const int k0 = ks << 5;

    // ---- stage A tile: 128 rows x 32 k, two 8-half chunks per thread ----
    #pragma unroll
    for (int e = 0; e < 2; ++e) {
      int c0  = tid*2 + e;            // 0..511
      int r   = c0 >> 2;
      int kk8 = (c0 & 3) << 3;
      int grow = rowBase + r;
      v8h hv;
      if (fullRows) {
        const float* src = X + (size_t)grow*Kpad + k0 + kk8;
        #pragma unroll
        for (int j = 0; j < 8; ++j) hv[j] = (_Float16)src[j];
      } else {
        // clamp row, two-phase (loads first, selects after) so loads pipeline
        int growc = grow < R ? grow : R - 1;
        const float* src = X + (size_t)growc*Kpad + k0 + kk8;
        float tmp[8];
        #pragma unroll
        for (int j = 0; j < 8; ++j) tmp[j] = src[j];
        #pragma unroll
        for (int j = 0; j < 8; ++j) hv[j] = (_Float16)(grow < R ? tmp[j] : 0.f);
      }
      *(v8h*)&As[r][kk8] = hv;
    }
    // ---- stage B tile transposed: 64 cols x 32 k, 8-half chunk/thread ----
    {
      int c  = tid & 63;
      int ko = (tid >> 6) << 3;       // 0,8,16,24
      int gc = colBase + c;           // Cout multiple of 64 -> in range
      v8h hv;
      #pragma unroll
      for (int j = 0; j < 8; ++j)
        hv[j] = (_Float16)W[(size_t)(k0 + ko + j)*Cout + gc];
      *(v8h*)&Bst[c][ko] = hv;
    }
    if (ks + 1 < Ksteps) {            // speculative prefetch of next A chunk
      __builtin_prefetch(X + (size_t)(rowBase + (tid >> 1))*Kpad + k0 + 32, 0, 1);
    }
    __syncthreads();

    // ---- fragments: K(i,lane) = (i&7) + koff + (i>=8 ? 16 : 0) ----
    const int koff = (lane >> 4) << 3;
    v16h aF[2], bF[2];
    #pragma unroll
    for (int t = 0; t < 2; ++t) {
      int rsub = wm*32 + t*16 + (lane & 15);
      v8h lo = *(const v8h*)&As[rsub][koff];
      v8h hi = *(const v8h*)&As[rsub][koff + 16];
      aF[t] = __builtin_shufflevector(lo, hi, 0,1,2,3,4,5,6,7,8,9,10,11,12,13,14,15);
    }
    #pragma unroll
    for (int u = 0; u < 2; ++u) {
      int csub = wn*32 + u*16 + (lane & 15);
      v8h lo = *(const v8h*)&Bst[csub][koff];
      v8h hi = *(const v8h*)&Bst[csub][koff + 16];
      bF[u] = __builtin_shufflevector(lo, hi, 0,1,2,3,4,5,6,7,8,9,10,11,12,13,14,15);
    }
    #pragma unroll
    for (int t = 0; t < 2; ++t)
      #pragma unroll
      for (int u = 0; u < 2; ++u)
        acc[t][u] = __builtin_amdgcn_wmma_f32_16x16x32_f16(false, aF[t], false, bF[u],
                                                           (short)0, acc[t][u], false, false);
    __syncthreads();
  }

  // ---- epilogue: bias + relu; C layout lane<16 -> M=r, lane>=16 -> M=r+8 ----
  #pragma unroll
  for (int u = 0; u < 2; ++u) {
    int col = colBase + wn*32 + u*16 + (lane & 15);
    float bv = bias[col];
    #pragma unroll
    for (int t = 0; t < 2; ++t) {
      int rbase = rowBase + wm*32 + t*16 + ((lane >> 4) << 3);
      #pragma unroll
      for (int r = 0; r < 8; ++r) {
        int grow = rbase + r;
        if (grow < R) {
          float v = acc[t][u][r] + bv;
          Y[(size_t)grow*Cout + col] = v > 0.f ? v : 0.f;
        }
      }
    }
  }
}

// ======================================================================
// BatchNorm (training-mode, biased stats, optional row mask)
// ======================================================================
__global__ void zero_kernel(float* __restrict__ p, long n) {
  long g = (long)blockIdx.x*blockDim.x + threadIdx.x;
  if (g < n) p[g] = 0.f;
}

__global__ void bn_stats(const float* __restrict__ X, const int* __restrict__ mask,
                         int rows, int C,
                         float* __restrict__ sum, float* __restrict__ sq,
                         float* __restrict__ cnt) {
  __shared__ float ls[1024];
  __shared__ float lq[1024];
  for (int c = threadIdx.x; c < C; c += blockDim.x) { ls[c] = 0.f; lq[c] = 0.f; }
  __syncthreads();
  int r0 = blockIdx.x * 256;
  int r1 = r0 + 256; if (r1 > rows) r1 = rows;
  float myc = 0.f;
  for (int r = r0; r < r1; ++r) {
    if (mask && !mask[r]) continue;
    const float* xr = X + (size_t)r*C;
    for (int c = threadIdx.x; c < C; c += blockDim.x) {
      float v = xr[c];
      atomicAdd(&ls[c], v);
      atomicAdd(&lq[c], v*v);
    }
    if (threadIdx.x == 0) myc += 1.f;
  }
  __syncthreads();
  for (int c = threadIdx.x; c < C; c += blockDim.x) {
    atomicAdd(&sum[c], ls[c]);
    atomicAdd(&sq[c],  lq[c]);
  }
  if (threadIdx.x == 0 && myc > 0.f) atomicAdd(cnt, myc);
}

__global__ void bn_apply(const float* __restrict__ X, float* __restrict__ Y,
                         const float* __restrict__ gam, const float* __restrict__ bet,
                         const float* __restrict__ sum, const float* __restrict__ sq,
                         const float* __restrict__ cnt, long total, int C) {
  long g = (long)blockIdx.x*blockDim.x + threadIdx.x;
  if (g >= total) return;
  int c = (int)(g % C);
  float n = *cnt;
  float m = sum[c] / n;
  float var = sq[c] / n - m*m;
  var = fmaxf(var, 0.f);
  float v = X[g];
  Y[g] = gam[c] * (v - m) * rsqrtf(var + EPSBN) + bet[c];
}

// ======================================================================
// masked max over neighbors, global max over centers
// ======================================================================
__global__ void max_agg(const float* __restrict__ X, const int* __restrict__ valid,
                        int centers, int C, float* __restrict__ out) {
  long g = (long)blockIdx.x*blockDim.x + threadIdx.x;
  long total = (long)centers * C;
  if (g >= total) return;
  long cen = g / C; int c = (int)(g % C);
  const float* xp = X + (cen*(size_t)KNBR)*C + c;
  const int*   vp = valid + cen*(size_t)KNBR;
  float best = -1e30f;
  for (int k = 0; k < KNBR; ++k)
    if (vp[k]) { float v = xp[(size_t)k*C]; best = v > best ? v : best; }
  out[g] = (best <= -1e30f) ? 0.f : best;
}

__global__ void global_max(const float* __restrict__ X, int Mrows, int C,
                           float* __restrict__ out, int total) {
  int g = blockIdx.x*blockDim.x + threadIdx.x;
  if (g >= total) return;
  int b = g / C, c = g % C;
  const float* xp = X + ((size_t)b*Mrows)*C + c;
  float best = -1e30f;
  for (int m = 0; m < Mrows; ++m) { float v = xp[(size_t)m*C]; best = v > best ? v : best; }
  out[g] = best;
}

// ======================================================================
// concats / kNN interpolation (all write Cpad-stride, zero-padded rows)
// ======================================================================
__global__ void concat_rows(const float* __restrict__ a, int Ca,
                            const float* __restrict__ b, int Cb,
                            float* __restrict__ out, long rows, int Cpad) {
  long total = rows * Cpad;
  long g = (long)blockIdx.x*blockDim.x + threadIdx.x;
  if (g >= total) return;
  long row = g / Cpad; int c = (int)(g % Cpad);
  float v = 0.f;
  if (c < Ca)           v = a[row*(size_t)Ca + c];
  else if (c < Ca + Cb) v = b[row*(size_t)Cb + (c - Ca)];
  out[g] = v;
}

__global__ void concat_bcast(const float* __restrict__ xg, int C1,
                             const float* __restrict__ xl, int C2,
                             float* __restrict__ out, int perB, long rows, int Cpad) {
  long total = rows * Cpad;
  long g = (long)blockIdx.x*blockDim.x + threadIdx.x;
  if (g >= total) return;
  long row = g / Cpad; int c = (int)(g % Cpad);
  int b = (int)(row / perB);
  float v = 0.f;
  if (c < C1)           v = xg[(size_t)b*C1 + c];
  else if (c < C1 + C2) v = xl[row*(size_t)C2 + (c - C1)];
  out[g] = v;
}

__global__ void knn3(const float* __restrict__ psrc, const float* __restrict__ ptgt,
                     int nsrc, int ntgt, int perB,
                     int* __restrict__ idx3, float* __restrict__ w3) {
  int g = blockIdx.x*blockDim.x + threadIdx.x;
  if (g >= ntgt) return;
  int b = g / perB;
  const float* P = psrc + (size_t)b*nsrc*3;
  float tx = ptgt[(size_t)g*3], ty = ptgt[(size_t)g*3+1], tz = ptgt[(size_t)g*3+2];
  float d0 = 1e30f, d1 = 1e30f, d2 = 1e30f;
  int   i0 = 0, i1 = 0, i2 = 0;
  for (int i = 0; i < nsrc; ++i) {
    float dx = P[i*3]-tx, dy = P[i*3+1]-ty, dz = P[i*3+2]-tz;
    float d = dx*dx + dy*dy + dz*dz;
    if (d < d0)      { d2 = d1; i2 = i1; d1 = d0; i1 = i0; d0 = d; i0 = i; }
    else if (d < d1) { d2 = d1; i2 = i1; d1 = d;  i1 = i; }
    else if (d < d2) { d2 = d;  i2 = i; }
  }
  idx3[(size_t)g*3+0] = i0; idx3[(size_t)g*3+1] = i1; idx3[(size_t)g*3+2] = i2;
  w3[(size_t)g*3+0] = 1.f / fmaxf(d0, 1e-16f);
  w3[(size_t)g*3+1] = 1.f / fmaxf(d1, 1e-16f);
  w3[(size_t)g*3+2] = 1.f / fmaxf(d2, 1e-16f);
}

__global__ void interp_concat(const float* __restrict__ hsrc, int Cs, int nsrc,
                              const int* __restrict__ idx3, const float* __restrict__ w3,
                              const float* __restrict__ xskip, int Ck,
                              int ntgt, int perB, int Cpad, float* __restrict__ out) {
  long total = (long)ntgt * Cpad;
  long g = (long)blockIdx.x*blockDim.x + threadIdx.x;
  if (g >= total) return;
  long row = g / Cpad; int c = (int)(g % Cpad);
  int b = (int)(row / perB);
  if (c < Cs) {
    float num = 0.f, den = 0.f;
    #pragma unroll
    for (int j = 0; j < 3; ++j) {
      int   s = idx3[row*3 + j];
      float w = w3[row*3 + j];
      num += w * hsrc[((size_t)b*nsrc + s)*Cs + c];
      den += w;
    }
    out[g] = num / den;
  } else if (c < Cs + Ck) {
    out[g] = xskip[row*(size_t)Ck + (c - Cs)];
  } else {
    out[g] = 0.f;
  }
}

// ======================================================================
// host driver
// ======================================================================
extern "C" void kernel_launch(void* const* d_in, const int* in_sizes, int n_in,
                              void* d_out, int out_size, void* d_ws, size_t ws_size,
                              hipStream_t stream) {
  (void)out_size; (void)ws_size;
  const float* x0   = (const float*)d_in[0];   // [B*N,4]
  const float* pos0 = (const float*)d_in[1];   // [B*N,3]

  // ---- param index layout: detect dict-insertion vs sorted-key flatten ----
  int sa1b[3], sa2b[3], sa3b[3], fp3b[2], fp2b[2], fp1b[3];
  bool orderB = (n_in >= 4 && in_sizes[3] == 7*64);   // sa1 first?
  int p = 3;
  if (orderB) {
    for (int i=0;i<3;i++){ sa1b[i]=p; p+=4; }
    for (int i=0;i<3;i++){ sa2b[i]=p; p+=4; }
    for (int i=0;i<3;i++){ sa3b[i]=p; p+=4; }
    for (int i=0;i<2;i++){ fp3b[i]=p; p+=4; }
    for (int i=0;i<2;i++){ fp2b[i]=p; p+=4; }
    for (int i=0;i<3;i++){ fp1b[i]=p; p+=4; }
  } else {
    for (int i=0;i<3;i++){ fp1b[i]=p; p+=4; }
    for (int i=0;i<2;i++){ fp2b[i]=p; p+=4; }
    for (int i=0;i<2;i++){ fp3b[i]=p; p+=4; }
    for (int i=0;i<3;i++){ sa1b[i]=p; p+=4; }
    for (int i=0;i<3;i++){ sa2b[i]=p; p+=4; }
    for (int i=0;i<3;i++){ sa3b[i]=p; p+=4; }
  }

  // ---- workspace carve-out ----
  char* ws = (char*)d_ws;
  size_t off = 0;
  auto take = [&](size_t bytes) -> void* {
    void* r = ws + off;
    off += (bytes + 255) & ~(size_t)255;
    return r;
  };
  float* bufA   = (float*)take((size_t)ROWS1*128*sizeof(float));
  float* bufB   = (float*)take((size_t)ROWS1*128*sizeof(float));
  int*   idx1   = (int*)  take((size_t)BB*MM1*sizeof(int));
  int*   idx2   = (int*)  take((size_t)BB*MM2*sizeof(int));
  float* pos1   = (float*)take((size_t)BB*MM1*3*sizeof(float));
  float* pos2   = (float*)take((size_t)BB*MM2*3*sizeof(float));
  int*   nidx1  = (int*)  take((size_t)ROWS1*sizeof(int));
  int*   valid1 = (int*)  take((size_t)ROWS1*sizeof(int));
  int*   nidx2  = (int*)  take((size_t)ROWS2*sizeof(int));
  int*   valid2 = (int*)  take((size_t)ROWS2*sizeof(int));
  float* x1     = (float*)take((size_t)BB*MM1*128*sizeof(float));
  float* x2     = (float*)take((size_t)BB*MM2*256*sizeof(float));
  float* x3     = (float*)take((size_t)BB*1024*sizeof(float));
  float* h3     = (float*)take((size_t)BB*MM2*256*sizeof(float));
  float* h2     = (float*)take((size_t)BB*MM1*128*sizeof(float));
  int*   idx3a  = (int*)  take((size_t)BB*MM1*3*sizeof(int));
  float* w3a    = (float*)take((size_t)BB*MM1*3*sizeof(float));
  int*   idx3b  = (int*)  take((size_t)BB*NN*3*sizeof(int));
  float* w3b    = (float*)take((size_t)BB*NN*3*sizeof(float));
  float* wpad   = (float*)take((size_t)512*1024*sizeof(float)); // max padded W
  float* stats  = (float*)take((size_t)2049*sizeof(float)); // sum[1024],sq[1024],cnt
  float* sSum = stats, *sSq = stats + 1024, *sCnt = stats + 2048;

  auto nb = [](long total, int t) -> unsigned { return (unsigned)((total + t - 1) / t); };

  // one MLP layer: (pad W if ragged) -> gemm(+bias,relu) -> bn stats -> bn apply
  auto run_layer = [&](const float* Xin, float* Yout, long rows, int Cin, int Cout,
                       int pb, const int* mask, float* normOut) {
    const float* Wp = (const float*)d_in[pb];
    const float* bp = (const float*)d_in[pb+1];
    const float* gp = (const float*)d_in[pb+2];
    const float* ep = (const float*)d_in[pb+3];
    int Kpad = (Cin + 31) & ~31;
    const float* Wuse = Wp;
    if (Kpad != Cin) {
      long wtot = (long)Kpad * Cout;
      pad_w<<<nb(wtot,256), 256, 0, stream>>>(Wp, wpad, Cin, Cout, wtot);
      Wuse = wpad;
    }
    dim3 grid(nb(rows, 128), (unsigned)(Cout/64));
    gemm_bias_relu_wmma<<<grid, 256, 0, stream>>>(Xin, Wuse, bp, Yout, (int)rows, Kpad, Cout);
    zero_kernel<<<nb(2049,256), 256, 0, stream>>>(stats, 2049);
    bn_stats<<<nb(rows,256), 256, 0, stream>>>(Yout, mask, (int)rows, Cout, sSum, sSq, sCnt);
    long tot = rows * (long)Cout;
    bn_apply<<<nb(tot,256), 256, 0, stream>>>(Yout, normOut ? normOut : Yout,
                                              gp, ep, sSum, sSq, sCnt, tot, Cout);
  };

  // ================= SA1 (msg Cin=7 -> pad 32) =================
  fps_kernel<<<BB, 256, 0, stream>>>(pos0, NN, MM1, idx1);
  gather_pos<<<nb(BB*MM1,256), 256, 0, stream>>>(pos0, idx1, NN, MM1, BB*MM1, pos1);
  radius_kernel<<<nb(BB*MM1,128), 128, 0, stream>>>(pos0, pos1, NN, MM1, RR1*RR1,
                                                    BB*MM1, nidx1, valid1);
  build_msg<<<nb(ROWS1,256), 256, 0, stream>>>(x0, pos0, pos1, nidx1, 4, NN, MM1,
                                               (long)ROWS1, 32, bufA);
  run_layer(bufA, bufB, ROWS1,   7,  64, sa1b[0], valid1, nullptr);
  run_layer(bufB, bufA, ROWS1,  64,  64, sa1b[1], valid1, nullptr);
  run_layer(bufA, bufB, ROWS1,  64, 128, sa1b[2], valid1, nullptr);
  max_agg<<<nb((long)BB*MM1*128,256), 256, 0, stream>>>(bufB, valid1, BB*MM1, 128, x1);

  // ================= SA2 (msg Cin=131 -> pad 160) =================
  fps_kernel<<<BB, 256, 0, stream>>>(pos1, MM1, MM2, idx2);
  gather_pos<<<nb(BB*MM2,256), 256, 0, stream>>>(pos1, idx2, MM1, MM2, BB*MM2, pos2);
  radius_kernel<<<nb(BB*MM2,128), 128, 0, stream>>>(pos1, pos2, MM1, MM2, RR2*RR2,
                                                    BB*MM2, nidx2, valid2);
  build_msg<<<nb(ROWS2,256), 256, 0, stream>>>(x1, pos1, pos2, nidx2, 128, MM1, MM2,
                                               (long)ROWS2, 160, bufA);
  run_layer(bufA, bufB, ROWS2, 131, 128, sa2b[0], valid2, nullptr);
  run_layer(bufB, bufA, ROWS2, 128, 128, sa2b[1], valid2, nullptr);
  run_layer(bufA, bufB, ROWS2, 128, 256, sa2b[2], valid2, nullptr);
  max_agg<<<nb((long)BB*MM2*256,256), 256, 0, stream>>>(bufB, valid2, BB*MM2, 256, x2);

  // ================= SA3 (concat Cin=259 -> pad 288) =================
  concat_rows<<<nb((long)BB*MM2*288,256), 256, 0, stream>>>(x2, 256, pos2, 3, bufA,
                                                            (long)BB*MM2, 288);
  run_layer(bufA, bufB, BB*MM2, 259,  256, sa3b[0], nullptr, nullptr);
  run_layer(bufB, bufA, BB*MM2, 256,  512, sa3b[1], nullptr, nullptr);
  run_layer(bufA, bufB, BB*MM2, 512, 1024, sa3b[2], nullptr, nullptr);
  global_max<<<nb(BB*1024,256), 256, 0, stream>>>(bufB, MM2, 1024, x3, BB*1024);

  // ================= FP3 (k=1 broadcast; Cin=1280 aligned) =================
  concat_bcast<<<nb((long)BB*MM2*1280,256), 256, 0, stream>>>(x3, 1024, x2, 256, bufA,
                                                              MM2, (long)BB*MM2, 1280);
  run_layer(bufA, bufB, BB*MM2, 1280, 256, fp3b[0], nullptr, nullptr);
  run_layer(bufB, h3,   BB*MM2,  256, 256, fp3b[1], nullptr, nullptr);

  // ================= FP2 (knn k=3: p2 -> p1; Cin=384 aligned) =================
  knn3<<<nb(BB*MM1,256), 256, 0, stream>>>(pos2, pos1, MM2, BB*MM1, MM1, idx3a, w3a);
  interp_concat<<<nb((long)BB*MM1*384,256), 256, 0, stream>>>(h3, 256, MM2, idx3a, w3a,
                                                              x1, 128, BB*MM1, MM1, 384, bufA);
  run_layer(bufA, bufB, BB*MM1, 384, 256, fp2b[0], nullptr, nullptr);
  run_layer(bufB, h2,   BB*MM1, 256, 128, fp2b[1], nullptr, nullptr);

  // ================= FP1 (knn k=3: p1 -> p0; Cin=132 -> pad 160) =================
  knn3<<<nb(BB*NN,256), 256, 0, stream>>>(pos1, pos0, MM1, BB*NN, NN, idx3b, w3b);
  interp_concat<<<nb((long)BB*NN*160,256), 256, 0, stream>>>(h2, 128, MM1, idx3b, w3b,
                                                             x0, 4, BB*NN, NN, 160, bufA);
  run_layer(bufA, bufB, BB*NN, 132, 128, fp1b[0], nullptr, nullptr);
  run_layer(bufB, bufA, BB*NN, 128, 128, fp1b[1], nullptr, nullptr);
  run_layer(bufA, bufB, BB*NN, 128, 128, fp1b[2], nullptr, (float*)d_out);
}
